// KnnEncoder_25271587570244
// MI455X (gfx1250) — compile-verified, matrix operand order checked by reading, not run on hardware
//
#include <hip/hip_runtime.h>

typedef __attribute__((ext_vector_type(16))) _Float16 v16h;
typedef __attribute__((ext_vector_type(8)))  _Float16 v8h;
typedef __attribute__((ext_vector_type(8)))  float    v8f;

#define B_GR  8
#define NP    4096
#define NPTS  (B_GR * NP)       // 32768
#define KNN   16
#define NEDGE (NPTS * KNN)      // 524288
#define BN_EPS 1e-5f
#define GWAVES 4                // waves per GEMM block (wave32) -> 64 rows/block

// ---------- order-preserving float <-> uint keys (for atomic max) ----------
__device__ __forceinline__ unsigned f2key(float f) {
  unsigned b = __float_as_uint(f);
  return b ^ ((unsigned)((int)b >> 31) | 0x80000000u);
}
__device__ __forceinline__ float key2f(unsigned k) {
  unsigned b = (k & 0x80000000u) ? (k ^ 0x80000000u) : ~k;
  return __uint_as_float(b);
}

// LDS byte address for async-to-LDS ops: generic shared pointers are
// {aperture_hi32, lds_offset32}; hardware uses addr[31:0] as the LDS address.
__device__ __forceinline__ unsigned lds_addr32(const void* p) {
  return (unsigned)(unsigned long long)p;
}

// async global->LDS copies (ASYNCcnt-tracked, CDNA5 ISA)
__device__ __forceinline__ void async_to_lds_b32(unsigned lds, const void* g) {
  asm volatile("global_load_async_to_lds_b32 %0, %1, off" :: "v"(lds), "v"(g) : "memory");
}
__device__ __forceinline__ void async_to_lds_b128(unsigned lds, const void* g) {
  asm volatile("global_load_async_to_lds_b128 %0, %1, off" :: "v"(lds), "v"(g) : "memory");
}
__device__ __forceinline__ void wait_asynccnt0() {
  asm volatile("s_wait_asynccnt 0x0" ::: "memory");
}

// ---------- zero init ----------
__global__ void zero_u32(unsigned* __restrict__ p, int n) {
  int i = blockIdx.x * blockDim.x + threadIdx.x;
  if (i < n) p[i] = 0u;
}

// ---------- per-graph kNN (one thread per query point, async LDS candidate tiles) ----
__global__ __launch_bounds__(256) void knn_kernel(const float* __restrict__ pos,
                                                  int* __restrict__ nbr) {
  __shared__ float sp[256 * 3];            // interleaved xyz candidate tile
  const int g  = blockIdx.x >> 4;          // 16 query tiles of 256 per graph
  const int ql = (blockIdx.x & 15) * 256 + threadIdx.x;
  const int q  = g * NP + ql;
  const float qx = pos[q * 3 + 0], qy = pos[q * 3 + 1], qz = pos[q * 3 + 2];

  float bd[KNN]; int bi[KNN];
#pragma unroll
  for (int t = 0; t < KNN; ++t) { bd[t] = 3.4e38f; bi[t] = 0x7fffffff; }

  for (int ct = 0; ct < NP / 256; ++ct) {
    __syncthreads();
    const float* src = pos + ((size_t)g * NP + ct * 256) * 3;
#pragma unroll
    for (int t = 0; t < 3; ++t)
      async_to_lds_b32(lds_addr32(&sp[threadIdx.x + t * 256]),
                       src + threadIdx.x + t * 256);
    wait_asynccnt0();                      // drain this wave's async copies
    __syncthreads();                       // then publish to the whole workgroup
    for (int j = 0; j < 256; ++j) {
      const float dx = sp[j * 3 + 0] - qx;
      const float dy = sp[j * 3 + 1] - qy;
      const float dz = sp[j * 3 + 2] - qz;
      const float d  = dx * dx + dy * dy + dz * dz;
      const int  idx = ct * 256 + j;
      if (d < bd[KNN - 1] || (d == bd[KNN - 1] && idx < bi[KNN - 1])) {
        int ip = KNN - 1;
#pragma unroll
        for (int t = KNN - 1; t > 0; --t) {
          const bool mv = (d < bd[t - 1]) || (d == bd[t - 1] && idx < bi[t - 1]);
          if (mv) { bd[t] = bd[t - 1]; bi[t] = bi[t - 1]; ip = t - 1; }
        }
#pragma unroll
        for (int t = 0; t < KNN; ++t)
          if (t == ip) { bd[t] = d; bi[t] = idx; }
      }
    }
  }
#pragma unroll
  for (int t = 0; t < KNN; ++t) nbr[q * KNN + t] = g * NP + bi[t];
}

// ---------- edge messages -> f16, padded 6 -> 32 ----------
__global__ void build_msg(const float* __restrict__ pos, const int* __restrict__ nbr,
                          _Float16* __restrict__ X0) {
  const int e = blockIdx.x * blockDim.x + threadIdx.x;
  if (e >= NEDGE) return;
  const int s = e / KNN;          // src = repeat(arange(N), K)
  const int d = nbr[e];           // dst = kNN neighbor
  const float sx = pos[s * 3], sy = pos[s * 3 + 1], sz = pos[s * 3 + 2];
  const float dx = pos[d * 3], dy = pos[d * 3 + 1], dz = pos[d * 3 + 2];
  _Float16* r = X0 + (size_t)e * 32;
  v8h v0 = { (_Float16)sx, (_Float16)sy, (_Float16)sz,
             (_Float16)(sx - dx), (_Float16)(sy - dy), (_Float16)(sz - dz),
             (_Float16)0.f, (_Float16)0.f };
  v8h z = {};
  *(v8h*)(r + 0)  = v0;
  *(v8h*)(r + 8)  = z;
  *(v8h*)(r + 16) = z;
  *(v8h*)(r + 24) = z;
}

// ---------- pack fp32 W[Kin,Nout] into WMMA B-fragment layout (f16) ----------
// block(nt,kt): 512 halves; half index = lane*16 + e, lane = hi*16 + n_lo
// k_local = (e<8) ? hi*8+e : 16+hi*8+(e-8)   (mirrors 16-bit A/B fragment layout)
__global__ void pack_w(const float* __restrict__ W, _Float16* __restrict__ Wp,
                       int Kin, int Kp, int Nout) {
  const int tid = blockIdx.x * blockDim.x + threadIdx.x;
  if (tid >= Kp * Nout) return;
  const int blk = tid >> 9, r = tid & 511;
  const int lane = r >> 4, e = r & 15;
  const int hi = lane >> 4, nn = lane & 15;
  const int kpt = Kp >> 5;
  const int kt = blk % kpt, nt = blk / kpt;
  const int kl = (e < 8) ? (hi * 8 + e) : (16 + hi * 8 + (e - 8));
  const int k  = kt * 32 + kl;
  const int n  = nt * 16 + nn;
  const float v = (k < Kin) ? W[k * Nout + n] : 0.f;
  Wp[tid] = (_Float16)v;
}

// ---------- fold BN stats into per-channel scale/shift ----------
__global__ void bn_finalize(const float* __restrict__ stats, const float* __restrict__ g,
                            const float* __restrict__ beta, float Minv, int Nout,
                            float* __restrict__ scale, float* __restrict__ shift) {
  const int j = blockIdx.x * blockDim.x + threadIdx.x;
  if (j >= Nout) return;
  const float mu  = stats[j] * Minv;
  const float var = stats[Nout + j] * Minv - mu * mu;
  const float sc  = g[j] * rsqrtf(var + BN_EPS);
  scale[j] = sc;
  shift[j] = beta[j] - mu * sc;
}

// ---------- generic WMMA GEMM: Y = act(X)*W + b, optional stats / atomic-max epilogues ----
__global__ __launch_bounds__(128) void gemm_wmma(
    const _Float16* __restrict__ X, int Kp,            // X: M x Kp (f16, Kp % 32 == 0)
    const _Float16* __restrict__ Wp,                   // packed B fragments
    const float* __restrict__ bias,                    // [Nout] or null
    const float* __restrict__ in_scale,                // folded BN scale for input, or null
    const float* __restrict__ in_shift,                // folded BN shift for input
    int Nout,
    _Float16* __restrict__ Yout,                       // store f16 M x Nout, or null
    float* __restrict__ stats,                         // [2*Nout] sum/sumsq atomics, or null
    unsigned* __restrict__ maxkey,                     // atomic-max key buffer, or null
    const int* __restrict__ dstidx,                    // per-row dst (featkey mode) or null
    int rows_per_seg)                                  // segment size (scene mode)
{
  extern __shared__ _Float16 smem[];
  const int lane = threadIdx.x & 31;
  const int wave = threadIdx.x >> 5;
  const int m    = lane & 15;
  const int hi   = lane >> 4;
  const int row0 = blockIdx.x * (GWAVES * 16) + wave * 16;
  _Float16* lds  = smem + (size_t)wave * 16 * Kp;      // private per-wave slice

  __builtin_prefetch(X + (size_t)row0 * Kp, 0, 0);

  // stage 16 rows into LDS
  const int vpr = Kp >> 3;                             // v8h vectors per row
  if (in_scale) {
    // apply previous layer's folded BN + ReLU on the fly (VALU path)
    for (int i = lane; i < 16 * vpr; i += 32) {
      const int r = i / vpr;
      const int c = (i % vpr) << 3;
      v8h x = *(const v8h*)(X + (size_t)(row0 + r) * Kp + c);
#pragma unroll
      for (int j = 0; j < 8; ++j) {
        const float f = (float)x[j] * in_scale[c + j] + in_shift[c + j];
        x[j] = (_Float16)fmaxf(f, 0.f);
      }
      *(v8h*)(lds + r * Kp + c) = x;
    }
    // same-wave LDS ops execute in order: fragment loads below see these stores
  } else {
    // identity input: pure copy -> async global->LDS, no VGPR round-trip
    for (int i = lane; i < 16 * vpr; i += 32) {
      const int r = i / vpr;
      const int c = (i % vpr) << 3;
      async_to_lds_b128(lds_addr32(lds + r * Kp + c),
                        X + (size_t)(row0 + r) * Kp + c);
    }
    wait_asynccnt0();   // wave-private slice: ASYNCcnt drain suffices, no barrier
  }

  const _Float16* myrow = lds + m * Kp;
  const int ktiles = Kp >> 5;

  int drow[8];
  if (maxkey) {
#pragma unroll
    for (int r8 = 0; r8 < 8; ++r8) {
      const int gr = row0 + r8 + 8 * hi;
      drow[r8] = dstidx ? dstidx[gr] : (gr / rows_per_seg);
    }
  }

  for (int nt = 0; nt < (Nout >> 4); ++nt) {
    v8f acc = {};
    const _Float16* wb = Wp + (size_t)(nt * ktiles) * 512 + lane * 16;
    for (int kt = 0; kt < ktiles; ++kt) {
      const v8h alo = *(const v8h*)(myrow + kt * 32 + hi * 8);
      const v8h ahi = *(const v8h*)(myrow + kt * 32 + 16 + hi * 8);
      v16h a;
#pragma unroll
      for (int j = 0; j < 8; ++j) { a[j] = alo[j]; a[8 + j] = ahi[j]; }
      const v16h b = *(const v16h*)(wb + (size_t)kt * 512);
      acc = __builtin_amdgcn_wmma_f32_16x16x32_f16(false, a, false, b,
                                                   (short)0, acc, false, false);
    }
    const int n = nt * 16 + m;
    const float bv = bias ? bias[n] : 0.f;
    float s = 0.f, ss = 0.f;
#pragma unroll
    for (int r8 = 0; r8 < 8; ++r8) {
      const float y = acc[r8] + bv;
      acc[r8] = y;
      s += y; ss += y * y;
    }
    if (stats) {
      s  += __shfl_xor(s, 16, 32);
      ss += __shfl_xor(ss, 16, 32);
      if (hi == 0) {
        atomicAdd(&stats[n], s);
        atomicAdd(&stats[Nout + n], ss);
      }
    }
    if (Yout) {
#pragma unroll
      for (int r8 = 0; r8 < 8; ++r8)
        Yout[(size_t)(row0 + r8 + 8 * hi) * Nout + n] = (_Float16)acc[r8];
    }
    if (maxkey) {
#pragma unroll
      for (int r8 = 0; r8 < 8; ++r8)
        atomicMax(&maxkey[(size_t)drow[r8] * Nout + n], f2key(acc[r8]));
    }
  }
}

// ---------- SA input: cat(feat[N,256], pos[N,3]) padded to 288, f16 ----------
__global__ void build_xsa(const unsigned* __restrict__ featkey, const float* __restrict__ pos,
                          _Float16* __restrict__ Xsa) {
  const int tid = blockIdx.x * blockDim.x + threadIdx.x;
  if (tid >= NPTS * 288) return;
  const int n = tid / 288, c = tid % 288;
  float v;
  if (c < 256)      v = key2f(featkey[(size_t)n * 256 + c]);
  else if (c < 259) v = pos[n * 3 + (c - 256)];
  else              v = 0.f;
  Xsa[tid] = (_Float16)v;
}

// ---------- output: cat(point_feat[B,256], scene_feat[B,512]) fp32 ----------
__global__ void finalize_out(const unsigned* __restrict__ featkey,
                             const unsigned* __restrict__ scenekey,
                             const int* __restrict__ qidx, float* __restrict__ out) {
  const int tid = blockIdx.x * blockDim.x + threadIdx.x;
  if (tid >= B_GR * 768) return;
  const int b = tid / 768, c = tid % 768;
  float v;
  if (c < 256) v = key2f(featkey[(size_t)qidx[b] * 256 + c]);
  else         v = key2f(scenekey[b * 512 + (c - 256)]);
  out[tid] = v;
}

extern "C" void kernel_launch(void* const* d_in, const int* in_sizes, int n_in,
                              void* d_out, int out_size, void* d_ws, size_t ws_size,
                              hipStream_t stream) {
  (void)in_sizes; (void)n_in; (void)out_size; (void)ws_size;
  // setup_inputs() dict order, lists flattened in insertion order:
  // 0:x 1:pos 2:batch 3:query_point_idx
  // conv: 4:W0 5:W1 6:W2 7:b0 8:b1 9:b2 10:g0 11:g1 12:beta0 13:beta1
  // sa:   14:W0 15:W1 16:W2 17:b0 18:b1 19:b2 20:g0 21:g1 22:beta0 23:beta1
  const float* pos  = (const float*)d_in[1];
  const int*   qidx = (const int*)d_in[3];
  const float* cW0 = (const float*)d_in[4],  *cW1 = (const float*)d_in[5],  *cW2 = (const float*)d_in[6];
  const float* cb0 = (const float*)d_in[7],  *cb1 = (const float*)d_in[8],  *cb2 = (const float*)d_in[9];
  const float* cg0 = (const float*)d_in[10], *cg1 = (const float*)d_in[11];
  const float* cB0 = (const float*)d_in[12], *cB1 = (const float*)d_in[13];
  const float* sW0 = (const float*)d_in[14], *sW1 = (const float*)d_in[15], *sW2 = (const float*)d_in[16];
  const float* sb0 = (const float*)d_in[17], *sb1 = (const float*)d_in[18], *sb2 = (const float*)d_in[19];
  const float* sg0 = (const float*)d_in[20], *sg1 = (const float*)d_in[21];
  const float* sB0 = (const float*)d_in[22], *sB1 = (const float*)d_in[23];

  // workspace bump allocator
  char* p = (char*)d_ws;
  auto alloc = [&](size_t bytes) -> char* {
    char* r = p; p += (bytes + 255) & ~(size_t)255; return r;
  };
  int*       nbr      = (int*)alloc((size_t)NEDGE * 4);
  _Float16*  X0       = (_Float16*)alloc((size_t)NEDGE * 32 * 2);
  _Float16*  Y1       = (_Float16*)alloc((size_t)NEDGE * 64 * 2);
  _Float16*  Y2       = (_Float16*)alloc((size_t)NEDGE * 128 * 2);
  unsigned*  featkey  = (unsigned*)alloc((size_t)NPTS * 256 * 4);
  _Float16*  Xsa      = (_Float16*)alloc((size_t)NPTS * 288 * 2);
  _Float16*  Ys1      = (_Float16*)alloc((size_t)NPTS * 256 * 2);
  _Float16*  Ys2      = (_Float16*)alloc((size_t)NPTS * 512 * 2);
  unsigned*  scenekey = (unsigned*)alloc((size_t)B_GR * 512 * 4);
  _Float16*  Wp0  = (_Float16*)alloc(32  * 64  * 2);
  _Float16*  Wp1  = (_Float16*)alloc(64  * 128 * 2);
  _Float16*  Wp2  = (_Float16*)alloc(128 * 256 * 2);
  _Float16*  Wps0 = (_Float16*)alloc(288 * 256 * 2);
  _Float16*  Wps1 = (_Float16*)alloc(256 * 512 * 2);
  _Float16*  Wps2 = (_Float16*)alloc(512 * 512 * 2);
  float* stats = (float*)alloc(1920 * 4);  // [2*64 | 2*128 | 2*256 | 2*512]
  float* st1 = stats, *st2 = stats + 128, *st3 = stats + 384, *st4 = stats + 896;
  float* bnbuf = (float*)alloc(1920 * 4);
  float* sc1 = bnbuf,        *sh1 = bnbuf + 64;
  float* sc2 = bnbuf + 128,  *sh2 = bnbuf + 256;
  float* sc3 = bnbuf + 384,  *sh3 = bnbuf + 640;
  float* sc4 = bnbuf + 896,  *sh4 = bnbuf + 1408;

  // re-initialize accumulators every call (graph-replay safe)
  zero_u32<<<(1920 + 255) / 256, 256, 0, stream>>>((unsigned*)stats, 1920);
  zero_u32<<<(NPTS * 256 + 255) / 256, 256, 0, stream>>>(featkey, NPTS * 256);
  zero_u32<<<(B_GR * 512 + 255) / 256, 256, 0, stream>>>(scenekey, B_GR * 512);

  // weight packing (fp32 -> f16 fragment layout)
  pack_w<<<(32  * 64  + 255) / 256, 256, 0, stream>>>(cW0, Wp0,  6,   32,  64);
  pack_w<<<(64  * 128 + 255) / 256, 256, 0, stream>>>(cW1, Wp1,  64,  64,  128);
  pack_w<<<(128 * 256 + 255) / 256, 256, 0, stream>>>(cW2, Wp2,  128, 128, 256);
  pack_w<<<(288 * 256 + 255) / 256, 256, 0, stream>>>(sW0, Wps0, 259, 288, 256);
  pack_w<<<(256 * 512 + 255) / 256, 256, 0, stream>>>(sW1, Wps1, 256, 256, 512);
  pack_w<<<(512 * 512 + 255) / 256, 256, 0, stream>>>(sW2, Wps2, 512, 512, 512);

  // kNN + edge messages
  knn_kernel<<<B_GR * (NP / 256), 256, 0, stream>>>(pos, nbr);
  build_msg<<<NEDGE / 256, 256, 0, stream>>>(pos, nbr, X0);

  const int EB = NEDGE / (GWAVES * 16);   // 8192 blocks
  const int NB = NPTS  / (GWAVES * 16);   // 512 blocks

  // conv MLP: 6->64 (BN stats), 64->128 (BN stats), 128->256 plain + segment_max
  gemm_wmma<<<EB, 128, 128 * 32,  stream>>>(X0, 32,  Wp0, cb0, nullptr, nullptr, 64,
                                            Y1, st1, nullptr, nullptr, 0);
  bn_finalize<<<1, 64, 0, stream>>>(st1, cg0, cB0, 1.f / NEDGE, 64, sc1, sh1);
  gemm_wmma<<<EB, 128, 128 * 64,  stream>>>(Y1, 64,  Wp1, cb1, sc1, sh1, 128,
                                            Y2, st2, nullptr, nullptr, 0);
  bn_finalize<<<1, 128, 0, stream>>>(st2, cg1, cB1, 1.f / NEDGE, 128, sc2, sh2);
  gemm_wmma<<<EB, 128, 128 * 128, stream>>>(Y2, 128, Wp2, cb2, sc2, sh2, 256,
                                            nullptr, nullptr, featkey, nbr, 0);

  // SA MLP: 259->256 (BN), 256->512 (BN), 512->512 plain + per-graph max
  build_xsa<<<(NPTS * 288 + 255) / 256, 256, 0, stream>>>(featkey, pos, Xsa);
  gemm_wmma<<<NB, 128, 128 * 288, stream>>>(Xsa, 288, Wps0, sb0, nullptr, nullptr, 256,
                                            Ys1, st3, nullptr, nullptr, 0);
  bn_finalize<<<1, 256, 0, stream>>>(st3, sg0, sB0, 1.f / NPTS, 256, sc3, sh3);
  gemm_wmma<<<NB, 128, 128 * 256, stream>>>(Ys1, 256, Wps1, sb1, sc3, sh3, 512,
                                            Ys2, st4, nullptr, nullptr, 0);
  bn_finalize<<<2, 256, 0, stream>>>(st4, sg1, sB1, 1.f / NPTS, 512, sc4, sh4);
  gemm_wmma<<<NB, 128, 128 * 512, stream>>>(Ys2, 512, Wps2, sb2, sc4, sh4, 512,
                                            nullptr, nullptr, scenekey, nullptr, NP);

  // gather query features + scene features
  finalize_out<<<(B_GR * 768 + 255) / 256, 256, 0, stream>>>(featkey, scenekey, qidx,
                                                             (float*)d_out);
}
